// MLP_01_56779467653695
// MI455X (gfx1250) — matrix operand order repaired and checked by reading, not compiled, not run on hardware
//
#include <hip/hip_runtime.h>
#include <hip/hip_bf16.h>
#include <stdint.h>

typedef __bf16 bf16_t;
typedef __attribute__((ext_vector_type(16))) __bf16 v16bf;
typedef __attribute__((ext_vector_type(8)))  float  v8f;

// ---------------------------------------------------------------------------
// Tiling configuration
//   block: 256 threads = 8 wave32s, arranged 4(M) x 2(N)
//   block tile: 256(M) x 128(N), K-step 64, 3-stage LDS pipeline (async DMA)
//   wave tile:  64(M) x 64(N)  -> 4x4 grid of 16x16 WMMA accumulators
//   => per 32-K step: 16 ds_load_b128 for 16 WMMAs (1.0 LDS op / WMMA)
//   Async loads complete in order per wave, so after issuing the 12 ops of
//   stage kt+2, s_wait_asynccnt <= 12 guarantees stage kt+1 is resident while
//   stage kt+2 stays in flight (2-deep DMA prefetch, no full drains).
// ---------------------------------------------------------------------------
#define BM   256
#define BN   128
#define BK   64
#define LDT  72          // LDS row stride in bf16 elements (BK + 8 pad)

// ---- async DMA: global -> LDS, 16 bytes per lane, tracked by ASYNCcnt ------
// LDS address operand = low 32 bits of the flat shared-pointer
// (ISA 10.2 aperture table).
__device__ inline void async_load_b128(const void* gp, void* lp) {
    const uint32_t lds32 = (uint32_t)(size_t)lp;
    const unsigned long long g64 = (unsigned long long)(size_t)gp;
    asm volatile("global_load_async_to_lds_b128 %0, %1, off"
                 :: "v"(lds32), "v"(g64)
                 : "memory");
}
__device__ inline void wait_async_le0()  { asm volatile("s_wait_asynccnt 0x0" ::: "memory"); }
__device__ inline void wait_async_le12() { asm volatile("s_wait_asynccnt 0xc" ::: "memory"); }

// ---- LDS fragment loaders (match gfx1250 WMMA 16-bit VGPR layouts) ---------
// A (16x32, MxK): lanes 0-15 -> K in [0,8)+[16,24); lanes 16-31 -> +8
__device__ inline v16bf load_frag_a(const bf16_t* lds, int row0, int kk, int lane) {
    const int m = lane & 15;
    const int h = lane >> 4;
    const bf16_t* p = lds + (row0 + m) * LDT + kk + h * 8;
    union { v16bf v; uint4 q[2]; } u;
    u.q[0] = *reinterpret_cast<const uint4*>(p);        // K = kb .. kb+7
    u.q[1] = *reinterpret_cast<const uint4*>(p + 16);   // K = kb+16 .. kb+23
    return u.v;
}
// B (32x16, KxN) with W stored transposed [N,K]: lanes 0-15 -> K in [0,8)+[16,24);
// lanes 16-31 -> +8 (same K order as A).
__device__ inline v16bf load_frag_b(const bf16_t* lds, int nrow0, int kk, int lane) {
    const int n = lane & 15;
    const int h = lane >> 4;
    const bf16_t* p = lds + (nrow0 + n) * LDT + kk + h * 8;
    union { v16bf v; uint4 q[2]; } u;
    u.q[0] = *reinterpret_cast<const uint4*>(p);
    u.q[1] = *reinterpret_cast<const uint4*>(p + 16);
    return u.v;
}

// ---------------------------------------------------------------------------
// GEMM:  OUT[M,N] = epilogue( A[M,K] * WT[N,K]^T )
//   MODE 0: out bf16 = bf16( max(acc*s, 0) * 2 )
//   MODE 1: out f32  = acc * s
// M divisible by 256; N divisible by 128; K divisible by 64.
// ---------------------------------------------------------------------------
template <int MODE>
__global__ __launch_bounds__(256)
void gemm_bf16_wmma(const bf16_t* __restrict__ A,
                    const bf16_t* __restrict__ WT,
                    void* __restrict__ OUT,
                    int K, int N,
                    const float* __restrict__ s_ptr)
{
    __shared__ bf16_t lds_a[3][BM * LDT];
    __shared__ bf16_t lds_b[3][BN * LDT];

    const int tid  = threadIdx.x;
    const int lane = tid & 31;
    const int wave = tid >> 5;
    const int wm   = wave >> 1;   // 0..3  (rows of 64)
    const int wn   = wave & 1;    // 0..1  (cols of 64)

    const int rowBase = blockIdx.y * BM;
    const int colBase = blockIdx.x * BN;
    const size_t Ks = (size_t)K;

    v8f acc[4][4];
    #pragma unroll
    for (int i = 0; i < 4; ++i)
        #pragma unroll
        for (int j = 0; j < 4; ++j)
            acc[i][j] = (v8f){0.f,0.f,0.f,0.f,0.f,0.f,0.f,0.f};

    // Per stage: each thread DMAs 8 x 16B chunks of A and 4 x 16B chunks of B
    // = 12 async ops (ASYNCcnt max 63; at most 24 in flight).
    auto issue_async = [&](int kt, int stage) {
        const size_t kof = (size_t)kt * BK;
        #pragma unroll
        for (int c = 0; c < 8; ++c) {           // A: 256 rows x 8 chunks
            const int cid = tid + (c << 8);
            const int r   = cid >> 3;           // 0..255
            const int k8  = (cid & 7) << 3;     // 0,8,...,56 (bf16 elements)
            async_load_b128(A + (size_t)(rowBase + r) * Ks + kof + k8,
                            &lds_a[stage][r * LDT + k8]);
        }
        #pragma unroll
        for (int c = 0; c < 4; ++c) {           // B: 128 rows x 8 chunks
            const int cid = tid + (c << 8);
            const int r   = cid >> 3;           // 0..127
            const int k8  = (cid & 7) << 3;
            async_load_b128(WT + (size_t)(colBase + r) * Ks + kof + k8,
                            &lds_b[stage][r * LDT + k8]);
        }
    };

    const int nk = K / BK;

    // Prologue: fill stages 0 and 1, then ensure stage 0 is resident.
    issue_async(0, 0);
    if (nk > 1) { issue_async(1, 1); wait_async_le12(); }
    else        { wait_async_le0(); }
    __syncthreads();

    int ct = 0;   // compute stage (kt % 3)
    int is = 2;   // issue stage ((kt+2) % 3)
    for (int kt = 0; kt < nk; ++kt) {
        const bool more = (kt + 2) < nk;
        if (more) issue_async(kt + 2, is);   // overwrite stage freed last iter

        #pragma unroll
        for (int ks = 0; ks < BK; ks += 32) {
            v16bf afrag[4], bfrag[4];
            #pragma unroll
            for (int i = 0; i < 4; ++i)
                afrag[i] = load_frag_a(lds_a[ct], wm * 64 + i * 16, ks, lane);
            #pragma unroll
            for (int j = 0; j < 4; ++j)
                bfrag[j] = load_frag_b(lds_b[ct], wn * 64 + j * 16, ks, lane);
            #pragma unroll
            for (int i = 0; i < 4; ++i)
                #pragma unroll
                for (int j = 0; j < 4; ++j)
                    acc[i][j] = __builtin_amdgcn_wmma_f32_16x16x32_bf16(
                        false, afrag[i], false, bfrag[j],
                        (short)0, acc[i][j], false, false);
        }

        if (kt + 1 < nk) {
            if (more) wait_async_le12();   // stage kt+1 landed; kt+2 in flight
            else      wait_async_le0();    // tail: nothing issued after kt+1
            __syncthreads();
        }
        ct = (ct == 2) ? 0 : ct + 1;
        is = (is == 2) ? 0 : is + 1;
    }

    // Epilogue.  D layout: element e of v8f -> m = (lane>=16)*8 + e, n = lane&15
    const float s = *s_ptr;
    const int mh = (lane >> 4) * 8;
    const int nl = lane & 15;
    if (MODE == 0) {
        bf16_t* out = reinterpret_cast<bf16_t*>(OUT);
        #pragma unroll
        for (int i = 0; i < 4; ++i)
            #pragma unroll
            for (int j = 0; j < 4; ++j)
                #pragma unroll
                for (int e = 0; e < 8; ++e) {
                    float v = acc[i][j][e] * s;
                    v = fmaxf(v, 0.0f) * 2.0f;
                    const int m = wm * 64 + i * 16 + mh + e;
                    const int n = wn * 64 + j * 16 + nl;
                    out[(size_t)(rowBase + m) * N + (colBase + n)] = (bf16_t)v;
                }
    } else {
        float* out = reinterpret_cast<float*>(OUT);
        #pragma unroll
        for (int i = 0; i < 4; ++i)
            #pragma unroll
            for (int j = 0; j < 4; ++j)
                #pragma unroll
                for (int e = 0; e < 8; ++e) {
                    const int m = wm * 64 + i * 16 + mh + e;
                    const int n = wn * 64 + j * 16 + nl;
                    out[(size_t)(rowBase + m) * N + (colBase + n)] = acc[i][j][e] * s;
                }
    }
}

// ---------------------------------------------------------------------------
// Preprocessing kernels
// ---------------------------------------------------------------------------
// x f32 -> bf16
__global__ void cvt_f32_bf16(const float* __restrict__ x, bf16_t* __restrict__ y, int n) {
    const int i = blockIdx.x * 256 + threadIdx.x;
    if (i < n) y[i] = (bf16_t)x[i];
}

// W0_eff^T [4096,1024]:  W0T[n][k] = 2*((int)k0[k][n] - (int)k0[k+1024][n])
// (folds the concat([x,-x]) into a {−2,0,+2} weight, halving layer-0 K)
__global__ void pack_w0(const uint8_t* __restrict__ k0, bf16_t* __restrict__ w0t) {
    const int idx = blockIdx.x * 256 + threadIdx.x;   // n*1024 + k
    const int k = idx & 1023;
    const int n = idx >> 10;
    const int a = (int)k0[(size_t)k * 4096 + n];
    const int b = (int)k0[(size_t)(k + 1024) * 4096 + n];
    w0t[idx] = (bf16_t)(float)(2 * (a - b));
}

// Generic transpose-pack: wt[n*K + k] = kk[k*N + n] ? +1 : -1
__global__ void pack_wT(const uint8_t* __restrict__ kk, bf16_t* __restrict__ wt,
                        int K, int N) {
    const int idx = blockIdx.x * 256 + threadIdx.x;
    const int k = idx % K;
    const int n = idx / K;
    wt[idx] = kk[(size_t)k * N + n] ? (bf16_t)1.0f : (bf16_t)(-1.0f);
}

// ---------------------------------------------------------------------------
extern "C" void kernel_launch(void* const* d_in, const int* in_sizes, int n_in,
                              void* d_out, int out_size, void* d_ws, size_t ws_size,
                              hipStream_t stream)
{
    const float*   x  = (const float*)d_in[0];
    const uint8_t* k0 = (const uint8_t*)d_in[1];
    const uint8_t* k1 = (const uint8_t*)d_in[2];
    const uint8_t* k2 = (const uint8_t*)d_in[3];
    const float*   s0 = (const float*)d_in[4];
    const float*   s1 = (const float*)d_in[5];
    const float*   s2 = (const float*)d_in[6];

    const int M  = 8 * 2048;   // 16384 tokens
    const int D  = 1024;
    const int F0 = 4096, F1 = 4096, F2 = 1024;

    char* ws = (char*)d_ws;
    bf16_t* Xb  = (bf16_t*)ws;  ws += (size_t)M  * D  * 2;
    bf16_t* W0T = (bf16_t*)ws;  ws += (size_t)F0 * D  * 2;
    bf16_t* W1T = (bf16_t*)ws;  ws += (size_t)F1 * F0 * 2;
    bf16_t* W2T = (bf16_t*)ws;  ws += (size_t)F2 * F1 * 2;
    bf16_t* H0  = (bf16_t*)ws;  ws += (size_t)M  * F0 * 2;
    bf16_t* H1  = (bf16_t*)ws;

    // preprocessing
    cvt_f32_bf16<<<(M * D) / 256, 256, 0, stream>>>(x, Xb, M * D);
    pack_w0    <<<(F0 * D) / 256, 256, 0, stream>>>(k0, W0T);
    pack_wT    <<<(F1 * F0) / 256, 256, 0, stream>>>(k1, W1T, F0, F1);
    pack_wT    <<<(F2 * F1) / 256, 256, 0, stream>>>(k2, W2T, F1, F2);

    // L0: H0 = relu(s0 * (Xb @ W0_eff)) * 2        [16384,1024]x[1024,4096]
    gemm_bf16_wmma<0><<<dim3(F0 / BN, M / BM), 256, 0, stream>>>(
        Xb, W0T, (void*)H0, D, F0, s0);
    // L1: H1 = relu(s1 * (H0 @ W1)) * 2            [16384,4096]x[4096,4096]
    gemm_bf16_wmma<0><<<dim3(F1 / BN, M / BM), 256, 0, stream>>>(
        H0, W1T, (void*)H1, F0, F1, s1);
    // L2: out = s2 * (H1 @ W2)  (f32)              [16384,4096]x[4096,1024]
    gemm_bf16_wmma<1><<<dim3(F2 / BN, M / BM), 256, 0, stream>>>(
        H1, W2T, d_out, F1, F2, s2);
}
